// Embedding_59476707115122
// MI455X (gfx1250) — matrix operand order repaired and checked by reading, not compiled, not run on hardware
//
#include <hip/hip_runtime.h>
#include <hip/hip_bf16.h>

// einsum("bvs,ve->bse"): per-batch GEMM  out[b] (S x E) = inputs[b]^T (S x V) * emb (V x E)
// B=128, V=2048, S=1024, E=64.  fp32 in/out; bf16-split WMMA (fp32 fidelity), f32 accumulate.
// Block tile: 128(s) x 64(e), 8 wave32s as 4(m) x 2(n), 2x2 16x16 tiles per wave.
// Register-staged, double-buffered LDS, one barrier per k-step.

typedef __attribute__((ext_vector_type(8)))  float  v8f;
typedef __attribute__((ext_vector_type(16))) __bf16 v16bf;
typedef __attribute__((ext_vector_type(8)))  __bf16 bf16x8;

#define B_   128
#define V_   2048
#define S_   1024
#define E_   64
#define KT   32          // k (vocab) tile per step
#define MT   128         // s rows per block
#define NKT  (V_ / KT)   // 64 k-steps
#define LDSPAD 40        // LDS row stride in bf16 elements (80B)

#define SHUF16(x, y) __builtin_shufflevector((x), (y), 0,1,2,3,4,5,6,7,8,9,10,11,12,13,14,15)

__global__ __launch_bounds__(256) void embed_gemm_wmma(const float* __restrict__ inp,
                                                       const float* __restrict__ emb,
                                                       float* __restrict__ out) {
  // double-buffered split-bf16 tiles, stored transposed ([m][k], [n][k]) for b128 fragments
  __shared__ __align__(16) __bf16 AsH[2][MT][LDSPAD];
  __shared__ __align__(16) __bf16 AsL[2][MT][LDSPAD];
  __shared__ __align__(16) __bf16 BsH[2][E_][LDSPAD];
  __shared__ __align__(16) __bf16 BsL[2][E_][LDSPAD];

  const int tid   = threadIdx.x;
  const int lane  = tid & 31;
  const int wv    = tid >> 5;    // 8 waves
  const int mw    = wv >> 1;     // 0..3 (32-row m slab)
  const int nw    = wv & 1;      // 0..1 (32-col n slab)
  const int khalf = lane >> 4;   // wave32 half select
  const int l15   = lane & 15;

  const int b  = blockIdx.x >> 3;          // 128 batches
  const int s0 = (blockIdx.x & 7) * MT;    // 8 s-tiles of 128

  // A staging: thread owns 16 consecutive k for one m (coalesced across lanes in s)
  const int m_a = tid & 127;
  const int k_a = (tid >> 7) * 16;         // 0 or 16
  // B staging: thread owns 8 consecutive k for one n (coalesced across lanes in e)
  const int n_b = tid & 63;
  const int k_b = (tid >> 6) * 8;          // 0,8,16,24

  const float* aBase = inp + (size_t)b * V_ * S_ + s0 + m_a;

  v8f c[2][2];
#pragma unroll
  for (int mi = 0; mi < 2; ++mi)
#pragma unroll
    for (int ni = 0; ni < 2; ++ni)
      c[mi][ni] = (v8f){0.f, 0.f, 0.f, 0.f, 0.f, 0.f, 0.f, 0.f};

  // ---- prologue: global-load k-tile 0 into registers
  float aReg[16], bReg[8];
#pragma unroll
  for (int j = 0; j < 16; ++j)
    aReg[j] = __builtin_nontemporal_load(aBase + (size_t)(k_a + j) * S_);
#pragma unroll
  for (int j = 0; j < 8; ++j)
    bReg[j] = emb[(size_t)(k_b + j) * E_ + n_b];

  int p = 0;
#pragma unroll 1
  for (int kt = 0; kt < NKT; ++kt) {
    // ---- convert regs -> split hi/lo bf16, store transposed into LDS[p]
#pragma unroll
    for (int ch = 0; ch < 2; ++ch) {
      bf16x8 h8, l8;
#pragma unroll
      for (int j = 0; j < 8; ++j) {
        float x = aReg[ch * 8 + j];
        __bf16 h = (__bf16)x;
        h8[j] = h;
        l8[j] = (__bf16)(x - (float)h);
      }
      *(bf16x8*)&AsH[p][m_a][k_a + ch * 8] = h8;   // ds_store_b128
      *(bf16x8*)&AsL[p][m_a][k_a + ch * 8] = l8;
    }
    {
      bf16x8 h8, l8;
#pragma unroll
      for (int j = 0; j < 8; ++j) {
        float x = bReg[j];
        __bf16 h = (__bf16)x;
        h8[j] = h;
        l8[j] = (__bf16)(x - (float)h);
      }
      *(bf16x8*)&BsH[p][n_b][k_b] = h8;
      *(bf16x8*)&BsL[p][n_b][k_b] = l8;
    }
    __syncthreads();

    // ---- kick off next k-tile's global loads (overlap with WMMA below)
    if (kt + 1 < NKT) {
      const int k0n = (kt + 1) * KT;
#pragma unroll
      for (int j = 0; j < 16; ++j)
        aReg[j] = __builtin_nontemporal_load(aBase + (size_t)(k0n + k_a + j) * S_);
#pragma unroll
      for (int j = 0; j < 8; ++j)
        bReg[j] = emb[(size_t)(k0n + k_b + j) * E_ + n_b];
    }

    // ---- fragments from LDS[p] (2x ds_load_b128 per matrix), 2x2 reuse
    v16bf aH[2], aL[2], bH[2], bL[2];
#pragma unroll
    for (int mi = 0; mi < 2; ++mi) {
      const int r = mw * 32 + mi * 16 + l15;
      bf16x8 t0 = *(const bf16x8*)&AsH[p][r][khalf * 8];
      bf16x8 t1 = *(const bf16x8*)&AsH[p][r][16 + khalf * 8];
      aH[mi] = SHUF16(t0, t1);
      bf16x8 u0 = *(const bf16x8*)&AsL[p][r][khalf * 8];
      bf16x8 u1 = *(const bf16x8*)&AsL[p][r][16 + khalf * 8];
      aL[mi] = SHUF16(u0, u1);
    }
#pragma unroll
    for (int ni = 0; ni < 2; ++ni) {
      const int r = nw * 32 + ni * 16 + l15;
      bf16x8 t0 = *(const bf16x8*)&BsH[p][r][khalf * 8];
      bf16x8 t1 = *(const bf16x8*)&BsH[p][r][16 + khalf * 8];
      bH[ni] = SHUF16(t0, t1);
      bf16x8 u0 = *(const bf16x8*)&BsL[p][r][khalf * 8];
      bf16x8 u1 = *(const bf16x8*)&BsL[p][r][16 + khalf * 8];
      bL[ni] = SHUF16(u0, u1);
    }

    // ---- 12 WMMAs: fp32-fidelity split  A*B ~= Ah*Bh + Ah*Bl + Al*Bh
#pragma unroll
    for (int mi = 0; mi < 2; ++mi)
#pragma unroll
      for (int ni = 0; ni < 2; ++ni) {
        c[mi][ni] = __builtin_amdgcn_wmma_f32_16x16x32_bf16(false, aH[mi], false, bH[ni],
                                                            (short)0, c[mi][ni], false, false);
        c[mi][ni] = __builtin_amdgcn_wmma_f32_16x16x32_bf16(false, aH[mi], false, bL[ni],
                                                            (short)0, c[mi][ni], false, false);
        c[mi][ni] = __builtin_amdgcn_wmma_f32_16x16x32_bf16(false, aL[mi], false, bH[ni],
                                                            (short)0, c[mi][ni], false, false);
      }

    p ^= 1;
  }

  // ---- store 2x2 16x16 f32 C tiles: VGPR r -> M = r + 8*khalf, N = lane&15
#pragma unroll
  for (int mi = 0; mi < 2; ++mi) {
#pragma unroll
    for (int ni = 0; ni < 2; ++ni) {
      const int row = s0 + mw * 32 + mi * 16 + khalf * 8;
      const int col = nw * 32 + ni * 16 + l15;
      float* o = out + ((size_t)b * S_ + row) * E_ + col;
#pragma unroll
      for (int r = 0; r < 8; ++r)
        o[(size_t)r * E_] = c[mi][ni][r];
    }
  }
}

extern "C" void kernel_launch(void* const* d_in, const int* in_sizes, int n_in,
                              void* d_out, int out_size, void* d_ws, size_t ws_size,
                              hipStream_t stream) {
  const float* inp = (const float*)d_in[0];   // [B, V, S] fp32
  const float* emb = (const float*)d_in[1];   // [V, E]   fp32
  float* out = (float*)d_out;                 // [B, S, E] fp32

  dim3 grid(B_ * (S_ / MT));   // 128 * 8 = 1024 blocks
  dim3 block(256);             // 8 wave32s
  embed_gemm_wmma<<<grid, block, 0, stream>>>(inp, emb, out);
}
